// MultiHeadAttention_69870527971624
// MI455X (gfx1250) — matrix operand order repaired and checked by reading, not compiled, not run on hardware
//
#include <hip/hip_runtime.h>
#include <cstdint>

#define NHEAD   8
#define M_NODES 20000
#define D_MODEL 512
#define N_EDGES 640000
#define DH      64

typedef __attribute__((ext_vector_type(16))) __bf16 v16bf;
typedef __attribute__((ext_vector_type(8)))  __bf16 v8bf;
typedef __attribute__((ext_vector_type(8)))  float  v8f;

// ---------------------------------------------------------------------------
// fp32 -> bf16 elementwise cast
// ---------------------------------------------------------------------------
__global__ void cast_f32_bf16_kernel(const float* __restrict__ in,
                                     __bf16* __restrict__ out, int n) {
  int i = blockIdx.x * blockDim.x + threadIdx.x;
  if (i < n) out[i] = (__bf16)in[i];
}

// ---------------------------------------------------------------------------
// OUT[M,N] = A[M,K](bf16) @ W[N,K](bf16)^T + bias[N]   (fp32 accumulate)
// One wave32 computes a 32x32 output tile as 2x2 WMMA sub-tiles
// (v_wmma_f32_16x16x32_bf16), sharing A- and B-fragments across the pairs:
// per K-step 8 b128 loads feed 4 WMMAs.
// A-frag per ISA layout: lanes 0-15 row m, K {0..7,16..23}; lanes 16-31 same
// rows, K {8..15,24..31}. B-frag: lane n holds the same K split of W[n,:].
// Requires Mrows % 32 == 0 and Ncols % 32 == 0.
// ---------------------------------------------------------------------------
__global__ void gemm_bf16_wmma_kernel(const __bf16* __restrict__ A,
                                      const __bf16* __restrict__ W,
                                      const float*  __restrict__ bias,
                                      float* __restrict__ OUT,
                                      int Mrows, int Ncols, int K) {
  const int wave = threadIdx.x >> 5;
  const int lane = threadIdx.x & 31;
  const int lrow = lane & 15;
  const int half = lane >> 4;
  const int tilesN = Ncols >> 5;            // 32-wide super-tiles in N

  const int tileId = blockIdx.x * (blockDim.x >> 5) + wave;
  const int tm = tileId / tilesN;
  const int tn = tileId % tilesN;
  if (tm * 32 >= Mrows) return;

  const __bf16* arow0 = A + (size_t)(tm * 32 + lrow) * K;
  const __bf16* arow1 = arow0 + (size_t)16 * K;
  const __bf16* wrow0 = W + (size_t)(tn * 32 + lrow) * K;
  const __bf16* wrow1 = wrow0 + (size_t)16 * K;

  v8f c00 = {}, c01 = {}, c10 = {}, c11 = {};
  for (int k0 = 0; k0 < K; k0 += 32) {
    union { v16bf v; v8bf h[2]; } a0, a1;
    a0.h[0] = *(const v8bf*)(arow0 + k0 + half * 8);
    a0.h[1] = *(const v8bf*)(arow0 + k0 + 16 + half * 8);
    a1.h[0] = *(const v8bf*)(arow1 + k0 + half * 8);
    a1.h[1] = *(const v8bf*)(arow1 + k0 + 16 + half * 8);
    union { v16bf v; v8bf h[2]; } bb0, bb1;
    bb0.h[0] = *(const v8bf*)(wrow0 + k0 + half * 8);
    bb0.h[1] = *(const v8bf*)(wrow0 + k0 + 16 + half * 8);
    bb1.h[0] = *(const v8bf*)(wrow1 + k0 + half * 8);
    bb1.h[1] = *(const v8bf*)(wrow1 + k0 + 16 + half * 8);
    v16bf b0 = bb0.v;
    v16bf b1 = bb1.v;
    c00 = __builtin_amdgcn_wmma_f32_16x16x32_bf16(false, a0.v, false, b0,
                                                  (short)0, c00, false, false);
    c01 = __builtin_amdgcn_wmma_f32_16x16x32_bf16(false, a0.v, false, b1,
                                                  (short)0, c01, false, false);
    c10 = __builtin_amdgcn_wmma_f32_16x16x32_bf16(false, a1.v, false, b0,
                                                  (short)0, c10, false, false);
    c11 = __builtin_amdgcn_wmma_f32_16x16x32_bf16(false, a1.v, false, b1,
                                                  (short)0, c11, false, false);
  }

  const float bv0 = bias[tn * 32 + lrow];
  const float bv1 = bias[tn * 32 + 16 + lrow];
  // D layout: VGPR r -> row (r + 8*half), column lane%16 within each 16x16.
  float* o00 = OUT + (size_t)(tm * 32 + 8 * half) * Ncols + tn * 32 + lrow;
  float* o10 = o00 + (size_t)16 * Ncols;
#pragma unroll
  for (int r = 0; r < 8; ++r) {
    o00[(size_t)r * Ncols]      = c00[r] + bv0;
    o00[(size_t)r * Ncols + 16] = c01[r] + bv1;
    o10[(size_t)r * Ncols]      = c10[r] + bv0;
    o10[(size_t)r * Ncols + 16] = c11[r] + bv1;
  }
}

// ---------------------------------------------------------------------------
// Ordered-uint mapping so float max can use integer atomicMax.
// ---------------------------------------------------------------------------
__device__ __forceinline__ unsigned ordf(float f) {
  unsigned u = __float_as_uint(f);
  return (u & 0x80000000u) ? ~u : (u | 0x80000000u);
}
__device__ __forceinline__ float unordf(unsigned u) {
  return (u & 0x80000000u) ? __uint_as_float(u & 0x7fffffffu)
                           : __uint_as_float(~u);
}

// ---------------------------------------------------------------------------
// Pass A: per-edge attention scores + segment max.
// One wave per edge; 4 lanes per head, 16 channels per lane.
// ---------------------------------------------------------------------------
__global__ void edge_scores_kernel(const long long* __restrict__ ei,
                                   const float* __restrict__ q,
                                   const float* __restrict__ kv,
                                   float* __restrict__ a,
                                   unsigned* __restrict__ amax) {
  const int e = blockIdx.x * (blockDim.x >> 5) + (threadIdx.x >> 5);
  if (e >= N_EDGES) return;
  const int lane = threadIdx.x & 31;
  const int h = lane >> 2;
  const int sub = lane & 3;

  const long long s = ei[e];
  const long long t = ei[N_EDGES + e];

  const float4* qv = (const float4*)(q  + (size_t)t * D_MODEL       + h * DH       + sub * 16);
  const float4* kk = (const float4*)(kv + (size_t)s * (2 * D_MODEL) + h * (2 * DH) + sub * 16);
  float p = 0.f;
#pragma unroll
  for (int j = 0; j < 4; ++j) {
    float4 x = qv[j], y = kk[j];
    p += x.x * y.x + x.y * y.y + x.z * y.z + x.w * y.w;
  }
  p += __shfl_xor(p, 1, 32);
  p += __shfl_xor(p, 2, 32);
  if (sub == 0) {
    const float val = p * 0.125f;  // 1/sqrt(64)
    a[(size_t)e * NHEAD + h] = val;
    atomicMax(&amax[(size_t)t * NHEAD + h], ordf(val));
  }
}

// ---------------------------------------------------------------------------
// Pass B: ex = exp(a - amax[tgt]); denom += ex; o[tgt] += ex * v[src]
// ---------------------------------------------------------------------------
__global__ void edge_softmax_accum_kernel(const long long* __restrict__ ei,
                                          const float* __restrict__ a,
                                          const unsigned* __restrict__ amax,
                                          const float* __restrict__ kv,
                                          float* __restrict__ denom,
                                          float* __restrict__ o) {
  const int e = blockIdx.x * (blockDim.x >> 5) + (threadIdx.x >> 5);
  if (e >= N_EDGES) return;
  const int lane = threadIdx.x & 31;
  const int h = lane >> 2;
  const int sub = lane & 3;

  const long long s = ei[e];
  const long long t = ei[N_EDGES + e];

  const float ex = __expf(a[(size_t)e * NHEAD + h] -
                          unordf(amax[(size_t)t * NHEAD + h]));
  if (sub == 0) atomicAdd(&denom[(size_t)t * NHEAD + h], ex);

  const float4* v4 = (const float4*)(kv + (size_t)s * (2 * D_MODEL) +
                                     h * (2 * DH) + DH + sub * 16);
  float* op = o + (size_t)t * D_MODEL + h * DH + sub * 16;
#pragma unroll
  for (int j = 0; j < 4; ++j) {
    float4 v = v4[j];
    atomicAdd(&op[4 * j + 0], ex * v.x);
    atomicAdd(&op[4 * j + 1], ex * v.y);
    atomicAdd(&op[4 * j + 2], ex * v.z);
    atomicAdd(&op[4 * j + 3], ex * v.w);
  }
}

// ---------------------------------------------------------------------------
// Pass C: o / denom, cast to bf16 for the output projection.
// ---------------------------------------------------------------------------
__global__ void normalize_cast_kernel(const float* __restrict__ o,
                                      const float* __restrict__ denom,
                                      __bf16* __restrict__ obf) {
  const size_t i = (size_t)blockIdx.x * blockDim.x + threadIdx.x;
  if (i >= (size_t)M_NODES * D_MODEL) return;
  const int m = (int)(i >> 9);
  const int c = (int)(i & 511);
  const int h = c >> 6;
  obf[i] = (__bf16)(o[i] / denom[(size_t)m * NHEAD + h]);
}

// ---------------------------------------------------------------------------
extern "C" void kernel_launch(void* const* d_in, const int* in_sizes, int n_in,
                              void* d_out, int out_size, void* d_ws, size_t ws_size,
                              hipStream_t stream) {
  const float*     s   = (const float*)d_in[0];
  const long long* ei  = (const long long*)d_in[1];
  const float*     Wq  = (const float*)d_in[2];
  const float*     bq  = (const float*)d_in[3];
  const float*     Wkv = (const float*)d_in[4];
  const float*     bkv = (const float*)d_in[5];
  const float*     Wo  = (const float*)d_in[6];
  const float*     bo  = (const float*)d_in[7];
  float* out = (float*)d_out;

  // ---- workspace layout (~229 MB) ----
  char* p = (char*)d_ws;
  auto alloc = [&](size_t bytes) -> char* {
    char* r = p;
    p += (bytes + 255) & ~(size_t)255;
    return r;
  };
  __bf16*   s_bf   = (__bf16*)  alloc((size_t)M_NODES * D_MODEL * 2);
  __bf16*   wq_bf  = (__bf16*)  alloc((size_t)D_MODEL * D_MODEL * 2);
  __bf16*   wkv_bf = (__bf16*)  alloc((size_t)2 * D_MODEL * D_MODEL * 2);
  __bf16*   wo_bf  = (__bf16*)  alloc((size_t)D_MODEL * D_MODEL * 2);
  float*    q      = (float*)   alloc((size_t)M_NODES * D_MODEL * 4);
  float*    kv     = (float*)   alloc((size_t)M_NODES * 2 * D_MODEL * 4);
  float*    a      = (float*)   alloc((size_t)N_EDGES * NHEAD * 4);
  unsigned* amax   = (unsigned*)alloc((size_t)M_NODES * NHEAD * 4);
  float*    denom  = (float*)   alloc((size_t)M_NODES * NHEAD * 4);
  float*    o      = (float*)   alloc((size_t)M_NODES * D_MODEL * 4);
  __bf16*   obf    = (__bf16*)  alloc((size_t)M_NODES * D_MODEL * 2);

  // ---- zero accumulators (amax=0 is the ordered-uint -inf) ----
  hipMemsetAsync(amax,  0, (size_t)M_NODES * NHEAD * 4, stream);
  hipMemsetAsync(denom, 0, (size_t)M_NODES * NHEAD * 4, stream);
  hipMemsetAsync(o,     0, (size_t)M_NODES * D_MODEL * 4, stream);

  // ---- fp32 -> bf16 casts ----
  {
    int n;
    n = M_NODES * D_MODEL;
    cast_f32_bf16_kernel<<<(n + 255) / 256, 256, 0, stream>>>(s, s_bf, n);
    n = D_MODEL * D_MODEL;
    cast_f32_bf16_kernel<<<(n + 255) / 256, 256, 0, stream>>>(Wq, wq_bf, n);
    n = 2 * D_MODEL * D_MODEL;
    cast_f32_bf16_kernel<<<(n + 255) / 256, 256, 0, stream>>>(Wkv, wkv_bf, n);
    n = D_MODEL * D_MODEL;
    cast_f32_bf16_kernel<<<(n + 255) / 256, 256, 0, stream>>>(Wo, wo_bf, n);
  }

  // ---- projections: q = s@Wq^T + bq ; kv = s@Wkv^T + bkv ----
  const int WPB = 4;              // waves per block (128 threads)
  {
    int tiles = (M_NODES / 32) * (D_MODEL / 32);           // 625*16 = 10000
    gemm_bf16_wmma_kernel<<<tiles / WPB, WPB * 32, 0, stream>>>(
        s_bf, wq_bf, bq, q, M_NODES, D_MODEL, D_MODEL);
  }
  {
    int tiles = (M_NODES / 32) * ((2 * D_MODEL) / 32);     // 625*32 = 20000
    gemm_bf16_wmma_kernel<<<tiles / WPB, WPB * 32, 0, stream>>>(
        s_bf, wkv_bf, bkv, kv, M_NODES, 2 * D_MODEL, D_MODEL);
  }

  // ---- edge passes (one wave32 per edge) ----
  {
    int blocks = N_EDGES / 8;     // 8 waves of 32 per 256-thread block
    edge_scores_kernel<<<blocks, 256, 0, stream>>>(ei, q, kv, a, amax);
    edge_softmax_accum_kernel<<<blocks, 256, 0, stream>>>(ei, a, amax, kv,
                                                          denom, o);
  }

  // ---- normalize + cast ----
  {
    size_t n = (size_t)M_NODES * D_MODEL;
    normalize_cast_kernel<<<(int)((n + 255) / 256), 256, 0, stream>>>(o, denom,
                                                                      obf);
  }

  // ---- output projection: out = obf@Wo^T + bo ----
  {
    int tiles = (M_NODES / 32) * (D_MODEL / 32);
    gemm_bf16_wmma_kernel<<<tiles / WPB, WPB * 32, 0, stream>>>(
        obf, wo_bf, bo, out, M_NODES, D_MODEL, D_MODEL);
  }
}
